// MultiHeadAttention_9388798509736
// MI455X (gfx1250) — matrix operand order repaired and checked by reading, not compiled, not run on hardware
//
#include <hip/hip_runtime.h>
#include <hip/hip_bf16.h>
#include <stdint.h>

#define BB 2
#define NN 512
#define MM 512
#define CC 256
#define HH 8
#define HC 32

typedef __attribute__((ext_vector_type(16))) __bf16 v16bf;
typedef __attribute__((ext_vector_type(8)))  float  v8f;
typedef __attribute__((ext_vector_type(4)))  float  v4f;

// ---- f32 -> bf16 WMMA operand builders -------------------------------------
// B operand (K x 16, 16-bit): lane = column, holds 16 consecutive K values.
// p must point at 16 consecutive floats, 16B-aligned.
__device__ __forceinline__ v16bf cvtB16(const float* __restrict__ p) {
    v4f x0 = ((const v4f*)p)[0];
    v4f x1 = ((const v4f*)p)[1];
    v4f x2 = ((const v4f*)p)[2];
    v4f x3 = ((const v4f*)p)[3];
    v16bf r;
    r[0] = (__bf16)x0[0]; r[1] = (__bf16)x0[1]; r[2]  = (__bf16)x0[2]; r[3]  = (__bf16)x0[3];
    r[4] = (__bf16)x1[0]; r[5] = (__bf16)x1[1]; r[6]  = (__bf16)x1[2]; r[7]  = (__bf16)x1[3];
    r[8] = (__bf16)x2[0]; r[9] = (__bf16)x2[1]; r[10] = (__bf16)x2[2]; r[11] = (__bf16)x2[3];
    r[12] = (__bf16)x3[0]; r[13] = (__bf16)x3[1]; r[14] = (__bf16)x3[2]; r[15] = (__bf16)x3[3];
    return r;
}

// Non-temporal variant for the one-shot 512 MiB embed_qk stream (TH=NT: keep
// the stream from sweeping L2 so V/K/Q/R/scores_e stay resident).
__device__ __forceinline__ v16bf cvtB16_nt(const float* __restrict__ p) {
    v4f x0 = __builtin_nontemporal_load((const v4f*)p);
    v4f x1 = __builtin_nontemporal_load(((const v4f*)p) + 1);
    v4f x2 = __builtin_nontemporal_load(((const v4f*)p) + 2);
    v4f x3 = __builtin_nontemporal_load(((const v4f*)p) + 3);
    v16bf r;
    r[0] = (__bf16)x0[0]; r[1] = (__bf16)x0[1]; r[2]  = (__bf16)x0[2]; r[3]  = (__bf16)x0[3];
    r[4] = (__bf16)x1[0]; r[5] = (__bf16)x1[1]; r[6]  = (__bf16)x1[2]; r[7]  = (__bf16)x1[3];
    r[8] = (__bf16)x2[0]; r[9] = (__bf16)x2[1]; r[10] = (__bf16)x2[2]; r[11] = (__bf16)x2[3];
    r[12] = (__bf16)x3[0]; r[13] = (__bf16)x3[1]; r[14] = (__bf16)x3[2]; r[15] = (__bf16)x3[3];
    return r;
}

// A operand (16 x K, 16-bit): lane holds its row's K run {p[0..15]}.
// Caller passes p = row + kstep*32 + ((lane>=16)?16:0); 16B-aligned.
__device__ __forceinline__ v16bf cvtA16(const float* __restrict__ p) {
    v4f x0 = ((const v4f*)p)[0];
    v4f x1 = ((const v4f*)p)[1];
    v4f x2 = ((const v4f*)p)[2];
    v4f x3 = ((const v4f*)p)[3];
    v16bf r;
    r[0] = (__bf16)x0[0]; r[1] = (__bf16)x0[1]; r[2]  = (__bf16)x0[2]; r[3]  = (__bf16)x0[3];
    r[4] = (__bf16)x1[0]; r[5] = (__bf16)x1[1]; r[6]  = (__bf16)x1[2]; r[7]  = (__bf16)x1[3];
    r[8] = (__bf16)x2[0]; r[9] = (__bf16)x2[1]; r[10] = (__bf16)x2[2]; r[11] = (__bf16)x2[3];
    r[12] = (__bf16)x3[0]; r[13] = (__bf16)x3[1]; r[14] = (__bf16)x3[2]; r[15] = (__bf16)x3[3];
    return r;
}

// ---- Kernel 1: Q/K/V projections  Y(1024x256) = X @ W^T + b  ----------------
// gridDim = (128, 3); 8 waves/block; one 16x16 output tile per wave.
__global__ __launch_bounds__(256) void proj_kernel(
    const float* __restrict__ Xq, const float* __restrict__ Xk, const float* __restrict__ Xv,
    const float* __restrict__ Wq, const float* __restrict__ Wk, const float* __restrict__ Wv,
    const float* __restrict__ bq, const float* __restrict__ bk, const float* __restrict__ bv,
    float* __restrict__ Yq, float* __restrict__ Yk, float* __restrict__ Yv)
{
    const float* X; const float* W; const float* bias; float* Y;
    if (blockIdx.y == 0)      { X = Xq; W = Wq; bias = bq; Y = Yq; }
    else if (blockIdx.y == 1) { X = Xk; W = Wk; bias = bk; Y = Yk; }
    else                      { X = Xv; W = Wv; bias = bv; Y = Yv; }

    const int lane = threadIdx.x & 31;
    const int wave = threadIdx.x >> 5;
    const int tile = blockIdx.x * 8 + wave;     // 0..1023
    const int rt = tile >> 4;                   // row tile 0..63
    const int ot = tile & 15;                   // col tile 0..15
    const int hi = (lane >= 16) ? 1 : 0;

    const int rrow = rt * 16 + (lane & 15);
    const int ocol = ot * 16 + (lane & 15);
    const float* xrow = X + (size_t)rrow * CC;
    const float* wrow = W + (size_t)ocol * CC;

    v8f acc = {};
#pragma unroll
    for (int kk = 0; kk < 8; ++kk) {
        v16bf a = cvtA16(xrow + kk * 32 + hi * 16);
        v16bf b = cvtB16(wrow + kk * 32 + hi * 16);
        acc = __builtin_amdgcn_wmma_f32_16x16x32_bf16(false, a, false, b,
                                                      (short)0, acc, false, false);
    }
    const float bo = bias[ocol];
    const int rbase = rt * 16 + hi * 8;
#pragma unroll
    for (int j = 0; j < 8; ++j)
        Y[(size_t)(rbase + j) * CC + ocol] = acc[j] + bo;
}

// ---- Kernel 2: R[bn][h][i] = sum_j Q[bn][h*32+j] * Wp[h*32+j][i] ------------
// 8192 waves: gw = h(8) x bnt(64) x it(16). One K-step WMMA each.
__global__ __launch_bounds__(256) void rproj_kernel(
    const float* __restrict__ Q, const float* __restrict__ Wp, float* __restrict__ R)
{
    const int lane = threadIdx.x & 31;
    const int gw = blockIdx.x * 8 + (threadIdx.x >> 5);
    const int h = gw >> 10;
    const int rem = gw & 1023;
    const int bnt = rem >> 4;
    const int it = rem & 15;
    const int hi = (lane >= 16) ? 1 : 0;

    const int bn = bnt * 16 + (lane & 15);
    const float* qrow = Q + (size_t)bn * CC + h * HC;
    v16bf a = cvtA16(qrow + hi * 16);

    const int icol = it * 16 + (lane & 15);
    const float* wcol = Wp + (size_t)(h * HC + hi * 16) * CC + icol;
    v16bf b;
#pragma unroll
    for (int j = 0; j < 16; ++j) b[j] = (__bf16)wcol[(size_t)j * CC];

    v8f acc = {};
    acc = __builtin_amdgcn_wmma_f32_16x16x32_bf16(false, a, false, b,
                                                  (short)0, acc, false, false);
    const int rbase = bnt * 16 + hi * 8;
#pragma unroll
    for (int j = 0; j < 8; ++j)
        R[((size_t)(rbase + j) * HH + h) * CC + icol] = acc[j];
}

// ---- Kernel 3: scores_e[b][h][n][m] = Q_h(512x32) @ K_h^T -------------------
// 16384 waves: gw = (b,h)(16) x nt(32) x mt(32). One K-step WMMA each.
__global__ __launch_bounds__(256) void se_kernel(
    const float* __restrict__ Q, const float* __restrict__ Kw, float* __restrict__ E)
{
    const int lane = threadIdx.x & 31;
    const int gw = blockIdx.x * 8 + (threadIdx.x >> 5);
    const int bh = gw >> 10;                    // 0..15
    const int b = bh >> 3, h = bh & 7;
    const int rem = gw & 1023;
    const int nt = rem >> 5;
    const int mt = rem & 31;
    const int hi = (lane >= 16) ? 1 : 0;

    const int nrow = nt * 16 + (lane & 15);
    const float* qrow = Q + (size_t)(b * NN + nrow) * CC + h * HC;
    v16bf a = cvtA16(qrow + hi * 16);

    const int mcol = mt * 16 + (lane & 15);
    const float* krow = Kw + (size_t)(b * MM + mcol) * CC + h * HC + hi * 16;
    v16bf bv = cvtB16(krow);

    v8f acc = {};
    acc = __builtin_amdgcn_wmma_f32_16x16x32_bf16(false, a, false, bv,
                                                  (short)0, acc, false, false);
    const int nbase = nt * 16 + hi * 8;
#pragma unroll
    for (int j = 0; j < 8; ++j)
        E[((size_t)(b * HH + h) * NN + (nbase + j)) * MM + mcol] = acc[j];
}

// ---- Kernel 4: fused scores_p + epilogue + softmax + attn@V -----------------
// One workgroup (8 waves) per (b,n). Streams embed_qk[b,n,:,:] exactly once
// with non-temporal loads (one-shot 512 MiB stream; keep L2 for V/K/E reuse).
__global__ __launch_bounds__(256) void attn_kernel(
    const float* __restrict__ Q, const float* __restrict__ V,
    const float* __restrict__ R, const float* __restrict__ Ews,
    const float* __restrict__ embed, const float* __restrict__ af,
    const float* __restrict__ kw, const uint8_t* __restrict__ mask,
    const float* __restrict__ bp, float* __restrict__ out)
{
    const int bn = blockIdx.x;                  // 0..1023
    const int b = bn >> 9, n = bn & 511;
    const int tid = threadIdx.x;
    const int wave = tid >> 5, lane = tid & 31;
    const int hi = (lane >= 16) ? 1 : 0;

    __shared__ float sc[HH][MM];                // 16 KB: scores -> attn
    __shared__ float s0[HH];

    // s0[h] = q_h . bp_h   (wave = head, shuffle reduce over 32 lanes)
    {
        float v = Q[(size_t)bn * CC + wave * HC + lane] * bp[wave * HC + lane];
#pragma unroll
        for (int off = 16; off > 0; off >>= 1) v += __shfl_xor(v, off, 32);
        if (lane == 0) s0[wave] = v;
    }

    // Hoist A operand: rows = heads (8 live of 16), K = 256 over 8 steps.
    v16bf Areg[8];
    {
        const int hr = lane & 15;
        const float* rrow = R + ((size_t)bn * HH + (hr & 7)) * CC;
#pragma unroll
        for (int kk = 0; kk < 8; ++kk) {
            v16bf a = cvtA16(rrow + kk * 32 + hi * 16);
            if (hr >= HH) {
#pragma unroll
                for (int i = 0; i < 16; ++i) a[i] = (__bf16)0.0f;
            }
            Areg[kk] = a;
        }
    }

    // scores_p: each wave owns 4 m-tiles; B operand streamed from embed_qk.
#pragma unroll
    for (int t = 0; t < 4; ++t) {
        const int m = (wave * 4 + t) * 16 + (lane & 15);
        const float* erow = embed + ((size_t)bn * MM + m) * CC;
        v8f acc = {};
#pragma unroll
        for (int kk = 0; kk < 8; ++kk) {
            v16bf bv = cvtB16_nt(erow + kk * 32 + hi * 16);
            acc = __builtin_amdgcn_wmma_f32_16x16x32_bf16(false, Areg[kk], false, bv,
                                                          (short)0, acc, false, false);
        }
        if (lane < 16) {
#pragma unroll
            for (int h = 0; h < HH; ++h) sc[h][m] = acc[h];
        }
    }
    __syncthreads();

    // epilogue: (s_p + s0 + s_e) * (1/sqrt(32)) * af[b,n,m] * kw[b,m], mask.
    const float inv = 0.17677669529663687f;     // 1/sqrt(32)
    for (int idx = tid; idx < HH * MM; idx += 256) {
        const int h = idx >> 9, m = idx & (MM - 1);
        float pre = sc[h][m] + s0[h]
                  + Ews[((size_t)(b * HH + h) * NN + n) * MM + m];
        pre = pre * inv * af[(size_t)bn * MM + m] * kw[b * MM + m];
        if (mask[b * MM + m]) pre = -__builtin_inff();
        sc[h][m] = pre;
    }
    __syncthreads();

    // softmax over m: wave = head.
    {
        const int h = wave;
        float mx = -__builtin_inff();
        for (int m = lane; m < MM; m += 32) mx = fmaxf(mx, sc[h][m]);
#pragma unroll
        for (int off = 16; off > 0; off >>= 1) mx = fmaxf(mx, __shfl_xor(mx, off, 32));
        float sum = 0.f;
        for (int m = lane; m < MM; m += 32) {
            float e = __expf(sc[h][m] - mx);
            sc[h][m] = e;
            sum += e;
        }
#pragma unroll
        for (int off = 16; off > 0; off >>= 1) sum += __shfl_xor(sum, off, 32);
        const float rs = 1.0f / sum;
        for (int m = lane; m < MM; m += 32) sc[h][m] *= rs;
    }
    __syncthreads();

    // hidden[b,n,h*32+c] = sum_m attn[h][m] * V[b,m,h*32+c]; wave=h, lane=c.
    {
        const int h = wave;
        const float* vb = V + (size_t)b * MM * CC + h * HC + lane;
        float acc = 0.f;
#pragma unroll 8
        for (int m = 0; m < MM; ++m) acc += sc[h][m] * vb[(size_t)m * CC];
        __builtin_nontemporal_store(acc, out + (size_t)bn * CC + h * HC + lane);
    }
}

// ---- Host launch ------------------------------------------------------------
extern "C" void kernel_launch(void* const* d_in, const int* in_sizes, int n_in,
                              void* d_out, int out_size, void* d_ws, size_t ws_size,
                              hipStream_t stream) {
    const float*   input_q = (const float*)d_in[0];
    const float*   input_k = (const float*)d_in[1];
    const float*   input_v = (const float*)d_in[2];
    const float*   embed   = (const float*)d_in[3];
    const float*   kweights= (const float*)d_in[4];
    const uint8_t* kmasks  = (const uint8_t*)d_in[5];
    const float*   af      = (const float*)d_in[6];
    const float*   Wq = (const float*)d_in[7];  const float* bq = (const float*)d_in[8];
    const float*   Wk = (const float*)d_in[9];  const float* bk = (const float*)d_in[10];
    const float*   Wv = (const float*)d_in[11]; const float* bv = (const float*)d_in[12];
    const float*   Wp = (const float*)d_in[13]; const float* bp = (const float*)d_in[14];
    float* out = (float*)d_out;

    // workspace layout (floats)
    float* ws = (float*)d_ws;
    float* Qws = ws;                            // 1024*256
    float* Kws = Qws + (size_t)1024 * CC;       // 1024*256
    float* Vws = Kws + (size_t)1024 * CC;       // 1024*256
    float* Rws = Vws + (size_t)1024 * CC;       // 1024*8*256
    float* Ews = Rws + (size_t)1024 * HH * CC;  // 2*8*512*512
    (void)in_sizes; (void)n_in; (void)out_size; (void)ws_size;

    proj_kernel<<<dim3(128, 3), 256, 0, stream>>>(
        input_q, input_k, input_v, Wq, Wk, Wv, bq, bk, bv, Qws, Kws, Vws);
    rproj_kernel<<<1024, 256, 0, stream>>>(Qws, Wp, Rws);
    se_kernel<<<2048, 256, 0, stream>>>(Qws, Kws, Ews);
    attn_kernel<<<1024, 256, 0, stream>>>(Qws, Vws, Rws, Ews, embed, af,
                                          kweights, kmasks, bp, out);
}
